// NonParamT2ISelfReferenceXFormersAttnProcessor_27187142984223
// MI455X (gfx1250) — matrix operand order repaired and checked by reading, not compiled, not run on hardware
//
#include <hip/hip_runtime.h>
#include <hip/hip_bf16.h>

typedef __attribute__((ext_vector_type(16))) __bf16 v16bf;
typedef __attribute__((ext_vector_type(8)))  float  v8f;

#define HEADS    8
#define NFRAMES  8
#define HW       1024
#define CDIM     640
#define DH       80
#define DHP      96          // dh padded to 3x32 for WMMA K-steps
#define KVLEN    2048
#define NTOK     (NFRAMES * HW)   // 8192 unique tokens

__device__ __forceinline__ unsigned short f32_bf16(float f) {
  unsigned u = __float_as_uint(f);
  unsigned r = u + 0x7FFFu + ((u >> 16) & 1u);   // round-to-nearest-even
  return (unsigned short)(r >> 16);
}

union FragBF { v16bf v; uint4 q[2]; };

// A fragment 16x32 (MxK) from row-major bf16, stride ld (elements).
// lane<16: M=lane, K=0..7 & 16..23 ; lane>=16: M=lane-16, K=8..15 & 24..31
__device__ __forceinline__ v16bf load_a(const unsigned short* base, int ld, int lane) {
  FragBF f;
  const unsigned short* p = base + (size_t)(lane & 15) * ld + ((lane >> 4) << 3);
  f.q[0] = *(const uint4*)p;
  f.q[1] = *(const uint4*)(p + 16);
  return f.v;
}

// B fragment 32x16 (KxN) loaded from N-major storage (row=N, col=K), stride ld.
// lane<16: N=lane, K=0..7 & 16..23 ; lane>=16: N=lane-16, K=8..15 & 24..31 (2 K per VGPR)
__device__ __forceinline__ v16bf load_b(const unsigned short* base, int ld, int lane) {
  FragBF f;
  const unsigned short* p = base + (size_t)(lane & 15) * ld + ((lane >> 4) << 3);
  f.q[0] = *(const uint4*)p;
  f.q[1] = *(const uint4*)(p + 16);
  return f.v;
}

// ---------------- conversion kernels ----------------
__global__ void cvt_x_kernel(const float* __restrict__ x, unsigned short* __restrict__ xb, int n) {
  int i = blockIdx.x * 256 + threadIdx.x;
  if (i < n) xb[i] = f32_bf16(x[i]);
}

// Wt[n*C + k] = bf16(W[k*C + n])  (N-major weights for B-fragment loads)
__global__ void cvt_wt_kernel(const float* __restrict__ w, unsigned short* __restrict__ wt) {
  int i = blockIdx.x * 256 + threadIdx.x;
  if (i < CDIM * CDIM) {
    int n = i / CDIM, k = i - n * CDIM;
    wt[i] = f32_bf16(w[k * CDIM + n]);
  }
}

// ---------------- WMMA GEMM: Y = A(8192x640) @ W(640x640) ----------------
// MODE 0: out bf16 (B*H, HW, DHP) row-major (Q/K, pad cols untouched -> pre-zeroed)
// MODE 1: out bf16 (B*H, DHP, HW) transposed (V)
// MODE 2: out f32 row-major 640 + bias (final projection)
template<int MODE>
__global__ __launch_bounds__(256) void gemm_wmma(
    const unsigned short* __restrict__ A,
    const unsigned short* __restrict__ Wt,
    unsigned short* __restrict__ outB,
    float* __restrict__ outF,
    const float* __restrict__ bias)
{
  const int lane = threadIdx.x & 31;
  const int w    = threadIdx.x >> 5;
  const int m0 = blockIdx.x * 128 + (w & 3) * 32;   // 4 waves along M
  const int n0 = blockIdx.y * 128 + (w >> 2) * 64;  // 2 waves along N

  v8f acc[2][4];
#pragma unroll
  for (int mi = 0; mi < 2; ++mi)
#pragma unroll
    for (int j = 0; j < 4; ++j)
#pragma unroll
      for (int r = 0; r < 8; ++r) acc[mi][j][r] = 0.0f;

  for (int k = 0; k < CDIM; k += 32) {
    v16bf a0 = load_a(A + (size_t)m0 * CDIM + k, CDIM, lane);
    v16bf a1 = load_a(A + (size_t)(m0 + 16) * CDIM + k, CDIM, lane);
#pragma unroll
    for (int j = 0; j < 4; ++j) {
      v16bf b = load_b(Wt + (size_t)(n0 + 16 * j) * CDIM + k, CDIM, lane);
      acc[0][j] = __builtin_amdgcn_wmma_f32_16x16x32_bf16(false, a0, false, b, (short)0, acc[0][j], false, false);
      acc[1][j] = __builtin_amdgcn_wmma_f32_16x16x32_bf16(false, a1, false, b, (short)0, acc[1][j], false, false);
    }
  }

#pragma unroll
  for (int mi = 0; mi < 2; ++mi) {
#pragma unroll
    for (int j = 0; j < 4; ++j) {
      int col = n0 + 16 * j + (lane & 15);
#pragma unroll
      for (int r = 0; r < 8; ++r) {
        int row = m0 + 16 * mi + r + ((lane >> 4) << 3);
        float v = acc[mi][j][r];
        if (MODE == 2) {
          outF[(size_t)row * CDIM + col] = v + bias[col];
        } else {
          int f = row >> 10;          // frame = row / HW
          int q = row & (HW - 1);
          int h = col / DH;
          int d = col - h * DH;       // 16-col tiles never straddle heads (80 = 5*16)
          size_t bh = (size_t)(f * HEADS + h);
          if (MODE == 0) outB[(bh * HW + q) * DHP + d] = f32_bf16(v);
          else           outB[(bh * DHP + d) * HW + q] = f32_bf16(v);
        }
      }
    }
  }
}

// ---------------- flash attention (online softmax, WMMA) ----------------
__global__ __launch_bounds__(128) void attn_wmma(
    const unsigned short* __restrict__ Qb,  // (B*H, HW, DHP)
    const unsigned short* __restrict__ Kb,  // (B*H, HW, DHP)
    const unsigned short* __restrict__ Vt,  // (B*H, DHP, HW)
    unsigned short* __restrict__ Ob,        // (NTOK, C) bf16
    const int* __restrict__ refIdx)
{
  __shared__ alignas(16) unsigned short pbuf[4][16 * 32];
  const int lane = threadIdx.x & 31;
  const int w    = threadIdx.x >> 5;
  const int bh = blockIdx.y;             // frame*HEADS + head
  const int f  = bh / HEADS;
  const int h  = bh - f * HEADS;
  const int q0 = blockIdx.x * 64 + w * 16;
  const int grp   = f / NFRAMES;         // batch group (b=1 -> 0)
  const int fref  = grp * NFRAMES + refIdx[0];
  const int bhref = fref * HEADS + h;

  const unsigned short* qbase = Qb + ((size_t)bh * HW + q0) * DHP;
  v16bf aq[3];
#pragma unroll
  for (int kc = 0; kc < 3; ++kc) aq[kc] = load_a(qbase + kc * 32, DHP, lane);

  v8f o[5];
  float m[8], l[8];
#pragma unroll
  for (int g = 0; g < 5; ++g)
#pragma unroll
    for (int r = 0; r < 8; ++r) o[g][r] = 0.0f;
#pragma unroll
  for (int r = 0; r < 8; ++r) { m[r] = -1e30f; l[r] = 0.0f; }

  const float scale = 0.11180339887498948f;   // 1/sqrt(80)
  const int Mh = (lane >> 4) << 3;
  const int N  = lane & 15;

  for (int j = 0; j < KVLEN; j += 32) {
    const int srcBH = (j < HW) ? bh : bhref;   // own tokens then reference tokens
    const int jj    = (j < HW) ? j : j - HW;
    const unsigned short* kbase = Kb + ((size_t)srcBH * HW + jj) * DHP;

    v8f s0, s1;
#pragma unroll
    for (int r = 0; r < 8; ++r) { s0[r] = 0.0f; s1[r] = 0.0f; }
#pragma unroll
    for (int kc = 0; kc < 3; ++kc) {
      v16bf bk0 = load_b(kbase + kc * 32, DHP, lane);
      v16bf bk1 = load_b(kbase + (size_t)16 * DHP + kc * 32, DHP, lane);
      s0 = __builtin_amdgcn_wmma_f32_16x16x32_bf16(false, aq[kc], false, bk0, (short)0, s0, false, false);
      s1 = __builtin_amdgcn_wmma_f32_16x16x32_bf16(false, aq[kc], false, bk1, (short)0, s1, false, false);
    }

    float mn[8];
#pragma unroll
    for (int r = 0; r < 8; ++r) {
      s0[r] *= scale; s1[r] *= scale;
      float mx = fmaxf(s0[r], s1[r]);            // row spans 16 lanes of this half-wave
      mx = fmaxf(mx, __shfl_xor(mx, 1, 32));
      mx = fmaxf(mx, __shfl_xor(mx, 2, 32));
      mx = fmaxf(mx, __shfl_xor(mx, 4, 32));
      mx = fmaxf(mx, __shfl_xor(mx, 8, 32));
      mn[r] = fmaxf(m[r], mx);
    }

#pragma unroll
    for (int r = 0; r < 8; ++r) {
      float corr = __expf(m[r] - mn[r]);
      float p0 = __expf(s0[r] - mn[r]);
      float p1 = __expf(s1[r] - mn[r]);
      float rs = p0 + p1;
      rs += __shfl_xor(rs, 1, 32);
      rs += __shfl_xor(rs, 2, 32);
      rs += __shfl_xor(rs, 4, 32);
      rs += __shfl_xor(rs, 8, 32);
      l[r] = l[r] * corr + rs;
      m[r] = mn[r];
#pragma unroll
      for (int g = 0; g < 5; ++g) o[g][r] *= corr;
      // C-layout (M=r+Mh, N) -> LDS, re-read as A-layout 16x32 fragment
      pbuf[w][(Mh + r) * 32 + N]      = f32_bf16(p0);
      pbuf[w][(Mh + r) * 32 + N + 16] = f32_bf16(p1);
    }
    asm volatile("s_wait_dscnt 0" ::: "memory");  // per-wave LDS RAW fence
    v16bf ap = load_a(pbuf[w], 32, lane);

    const unsigned short* vbase = Vt + (size_t)srcBH * DHP * HW + jj;
#pragma unroll
    for (int g = 0; g < 5; ++g) {                 // dh groups 0..79; group 5 is pad
      v16bf bv = load_b(vbase + (size_t)(16 * g) * HW, HW, lane);
      o[g] = __builtin_amdgcn_wmma_f32_16x16x32_bf16(false, ap, false, bv, (short)0, o[g], false, false);
    }
  }

#pragma unroll
  for (int g = 0; g < 5; ++g) {
    int col = h * DH + 16 * g + N;
#pragma unroll
    for (int r = 0; r < 8; ++r) {
      int row = f * HW + q0 + r + Mh;
      Ob[(size_t)row * CDIM + col] = f32_bf16(o[g][r] / l[r]);
    }
  }
}

// ---------------- host side ----------------
extern "C" void kernel_launch(void* const* d_in, const int* in_sizes, int n_in,
                              void* d_out, int out_size, void* d_ws, size_t ws_size,
                              hipStream_t stream) {
  const float* hs  = (const float*)d_in[0];
  const float* Wq  = (const float*)d_in[1];
  const float* Wk  = (const float*)d_in[2];
  const float* Wv  = (const float*)d_in[3];
  const float* Wo  = (const float*)d_in[4];
  const float* bo  = (const float*)d_in[5];
  const int*   idx = (const int*)d_in[6];
  float* out = (float*)d_out;

  // workspace carve-up (bf16 buffers), 256B aligned
  char* ws = (char*)d_ws;
  size_t off = 0;
  auto carve = [&](size_t bytes) -> char* {
    char* p = ws + off;
    off = (off + bytes + 255) & ~(size_t)255;
    return p;
  };
  unsigned short* Xb  = (unsigned short*)carve((size_t)NTOK * CDIM * 2);
  unsigned short* Wtq = (unsigned short*)carve((size_t)CDIM * CDIM * 2);
  unsigned short* Wtk = (unsigned short*)carve((size_t)CDIM * CDIM * 2);
  unsigned short* Wtv = (unsigned short*)carve((size_t)CDIM * CDIM * 2);
  unsigned short* Wto = (unsigned short*)carve((size_t)CDIM * CDIM * 2);
  const size_t qkvElems = (size_t)NFRAMES * HEADS * HW * DHP;
  unsigned short* Qb = (unsigned short*)carve(qkvElems * 2);
  unsigned short* Kb = (unsigned short*)carve(qkvElems * 2);
  unsigned short* Vt = (unsigned short*)carve(qkvElems * 2);
  unsigned short* Ob = (unsigned short*)carve((size_t)NTOK * CDIM * 2);

  // 1) bf16 conversions
  cvt_x_kernel<<<(NTOK * CDIM + 255) / 256, 256, 0, stream>>>(hs, Xb, NTOK * CDIM);
  int wn = (CDIM * CDIM + 255) / 256;
  cvt_wt_kernel<<<wn, 256, 0, stream>>>(Wq, Wtq);
  cvt_wt_kernel<<<wn, 256, 0, stream>>>(Wk, Wtk);
  cvt_wt_kernel<<<wn, 256, 0, stream>>>(Wv, Wtv);
  cvt_wt_kernel<<<wn, 256, 0, stream>>>(Wo, Wto);

  // 2) zero Q/K/V pads (dh 80..95 must be 0 so padded WMMA K-steps are exact)
  hipMemsetAsync(Qb, 0, qkvElems * 2, stream);
  hipMemsetAsync(Kb, 0, qkvElems * 2, stream);
  hipMemsetAsync(Vt, 0, qkvElems * 2, stream);

  // 3) projections
  dim3 gb(NTOK / 128, CDIM / 128);
  gemm_wmma<0><<<gb, 256, 0, stream>>>(Xb, Wtq, Qb, nullptr, nullptr);
  gemm_wmma<0><<<gb, 256, 0, stream>>>(Xb, Wtk, Kb, nullptr, nullptr);
  gemm_wmma<1><<<gb, 256, 0, stream>>>(Xb, Wtv, Vt, nullptr, nullptr);

  // 4) attention (64 q-rows per block, 4 waves)
  attn_wmma<<<dim3(HW / 64, NFRAMES * HEADS), 128, 0, stream>>>(Qb, Kb, Vt, Ob, idx);

  // 5) output projection + bias (f32 out)
  gemm_wmma<2><<<gb, 256, 0, stream>>>(Ob, Wto, nullptr, out, bo);

  (void)in_sizes; (void)n_in; (void)out_size; (void)ws_size;
}
